// LogicalHead_23613730194036
// MI455X (gfx1250) — matrix operand order repaired and checked by reading, not compiled, not run on hardware
//
#include <hip/hip_runtime.h>
#include <hip/hip_fp16.h>

// ---------------------------------------------------------------------------
// Graph attention pooling head for MI455X (gfx1250, wave32, WMMA).
//   score = tanh(h @ gW1 + gb1) @ gw2 + gb2        (N x 1)
//   attn  = segment_softmax(score)                  per sorted segment
//   g     = [segsum(h*attn), segmax(h)]             (B x 2H)
//   out   = relu(g @ mW1 + mb1) @ mw2 + mb2         (B x 1)
// Strategy: stream h once (f32 in HBM), convert tiles to f16 in LDS, do all
// GEMMs with v_wmma_f32_16x16x32_f16 (f32 accum). Pooling re-read is L2-hot.
// tanh uses the native gfx1250 V_TANH_F32 transcendental (libm tanhf costs
// ~30 branchy VALU ops x 134M evals, which would dominate the HBM roofline).
// ---------------------------------------------------------------------------

typedef __attribute__((ext_vector_type(16))) _Float16 v16h;
typedef __attribute__((ext_vector_type(8)))  _Float16 v8h;
typedef __attribute__((ext_vector_type(8)))  float    v8f;

#define HDIM   128
#define MAXSEG 4096
#define NWAVES 8
#define STRIDE 136   // padded halves per staged row (avoids LDS bank conflicts)

__device__ __forceinline__ v16h load_frag(const _Float16* p) {
  // two 16-byte loads -> 16 halves
  v8h lo = *(const v8h*)p;
  v8h hi = *(const v8h*)(p + 16);
  return __builtin_shufflevector(lo, hi, 0,1,2,3,4,5,6,7,8,9,10,11,12,13,14,15);
}

__device__ __forceinline__ float group_sum16(float v) {
  // butterfly over 16-lane halves of the wave (masks < 16 keep groups intact)
  v += __shfl_xor(v, 1, 32);
  v += __shfl_xor(v, 2, 32);
  v += __shfl_xor(v, 4, 32);
  v += __shfl_xor(v, 8, 32);
  return v;
}

__device__ __forceinline__ float tanh_fast(float x) {
#if __has_builtin(__builtin_amdgcn_tanhf)
  return __builtin_amdgcn_tanhf(x);        // native V_TANH_F32, hazards by clang
#else
  float r;
  // ISA: 1 independent op or V_NOP required after a TRANS op before use
  asm("v_tanh_f32 %0, %1\n\tv_nop" : "=v"(r) : "v"(x));
  return r;
#endif
}

// ---------------------------------------------------------------------------
// Kernel 1: pre-swizzle gate_w1 (128x128) and mlp_w1 (256x128) into f16
// B-fragment layout.  B layout (16x16x32 f16): lane l (col N = l&15) holds
// K = (l<16 ? 0 : 8) + (i<8 ? i : i+8), i=0..15 (matches A table order).
// gate frag index f = nt*4 + ks (ks: K/32 slice).  mlp frag f = nt*8 + ks.
// ---------------------------------------------------------------------------
__global__ void prep_frags(const float* __restrict__ gw1,
                           const float* __restrict__ mw1,
                           _Float16* __restrict__ gateF,
                           _Float16* __restrict__ mlpF) {
  int t = blockIdx.x * blockDim.x + threadIdx.x;
  if (t < 32 * 32) {                       // 32 gate fragments
    int f = t >> 5, lane = t & 31;
    int nt = f >> 2, ks = f & 3;
    int kbase = ks * 32 + ((lane < 16) ? 0 : 8);
    int n = nt * 16 + (lane & 15);
    _Float16* dst = gateF + ((size_t)(f * 32 + lane)) * 16;
#pragma unroll
    for (int i = 0; i < 16; ++i) dst[i] = (_Float16)gw1[(kbase + (i < 8 ? i : i + 8)) * HDIM + n];
  } else if (t < 32 * 32 + 64 * 32) {      // 64 mlp fragments
    int idx = t - 32 * 32;
    int f = idx >> 5, lane = idx & 31;
    int nt = f >> 3, ks = f & 7;
    int kbase = ks * 32 + ((lane < 16) ? 0 : 8);
    int n = nt * 16 + (lane & 15);
    _Float16* dst = mlpF + ((size_t)(f * 32 + lane)) * 16;
#pragma unroll
    for (int i = 0; i < 16; ++i) dst[i] = (_Float16)mw1[(kbase + (i < 8 ? i : i + 8)) * HDIM + n];
  }
}

// ---------------------------------------------------------------------------
// Kernel 2: one workgroup per segment. Phases: gate WMMA scores -> softmax ->
// pooling. Writes g row as f16 (256 halves) for the MLP kernel.
// ---------------------------------------------------------------------------
__global__ __launch_bounds__(256, 2) void seg_kernel(
    const float* __restrict__ h, const int* __restrict__ batch,
    const float* __restrict__ gb1, const float* __restrict__ gw2,
    const float* __restrict__ gb2, const _Float16* __restrict__ gateF,
    _Float16* __restrict__ g16, int N) {
  __shared__ __align__(16) _Float16 stage[NWAVES][16 * STRIDE];
  __shared__ float sc[MAXSEG];
  __shared__ float red[256];
  __shared__ float pa[128], pm[128];
  __shared__ int bounds[2];

  const int tid = threadIdx.x;
  const int b = blockIdx.x;

  if (tid == 0) {  // binary-search segment boundaries in the sorted batch array
    int lo = 0, hi = N;
    while (lo < hi) { int mid = (lo + hi) >> 1; if (batch[mid] < b) lo = mid + 1; else hi = mid; }
    bounds[0] = lo;
    int lo2 = lo, hi2 = N;
    while (lo2 < hi2) { int mid = (lo2 + hi2) >> 1; if (batch[mid] < b + 1) lo2 = mid + 1; else hi2 = mid; }
    bounds[1] = lo2;
  }
  __syncthreads();
  const int n0 = bounds[0];
  int len = bounds[1] - n0;
  if (len > MAXSEG) len = MAXSEG;
  const int nch = (len + 15) >> 4;

  const int wave = tid >> 5, lane = tid & 31;
  const int lanelo = lane & 15;
  _Float16* st = stage[wave];

  // ---- Phase 1: gate scores, 16-node chunks round-robined over 8 waves ----
  for (int ch = wave; ch < nch; ch += NWAVES) {
#pragma unroll 4
    for (int r = 0; r < 16; ++r) {               // stage rows f32 -> f16 LDS
      int node = ch * 16 + r;
      float4 v = make_float4(0.f, 0.f, 0.f, 0.f);
      if (node < len) {
        v = *(const float4*)(h + (size_t)(n0 + node) * HDIM + lane * 4);
        int pn = n0 + node + 16;                 // prefetch next chunk
        if (pn < N) __builtin_prefetch(h + (size_t)pn * HDIM + lane * 4, 0, 1);
      }
      union { _Float16 q[4]; uint2 u; } pk;
      pk.q[0] = (_Float16)v.x; pk.q[1] = (_Float16)v.y;
      pk.q[2] = (_Float16)v.z; pk.q[3] = (_Float16)v.w;
      *(uint2*)&st[r * STRIDE + lane * 4] = pk.u;
    }
    asm volatile("s_wait_dscnt 0" ::: "memory"); // intra-wave LDS RAW

    // A fragments (ISA 16-bit A layout), reused across all 8 N-tiles
    const int koff = (lane < 16) ? 0 : 8;
    v16h A[4];
#pragma unroll
    for (int ks = 0; ks < 4; ++ks) {
      const _Float16* p = &st[lanelo * STRIDE + ks * 32 + koff];
      A[ks] = load_frag(p);
    }

    float part[8];                               // per-lane partials (reduce once)
#pragma unroll
    for (int r = 0; r < 8; ++r) part[r] = 0.f;

#pragma unroll
    for (int nt = 0; nt < 8; ++nt) {
      float bias = gb1[nt * 16 + lanelo];        // C layout: N = lane & 15
      v8f c;
#pragma unroll
      for (int r = 0; r < 8; ++r) c[r] = bias;
#pragma unroll
      for (int ks = 0; ks < 4; ++ks) {
        v16h Bf = *(const v16h*)(gateF + ((size_t)((nt * 4 + ks) * 32 + lane)) * 16);
        c = __builtin_amdgcn_wmma_f32_16x16x32_f16(false, A[ks], false, Bf,
                                                   (short)0, c, false, false);
      }
      float w2c = gw2[nt * 16 + lanelo];
#pragma unroll
      for (int r = 0; r < 8; ++r)
        part[r] += tanh_fast(c[r]) * w2c;        // native V_TANH_F32
    }
    float bb2 = gb2[0];
    float scc[8];
#pragma unroll
    for (int r = 0; r < 8; ++r)
      scc[r] = group_sum16(part[r]);             // one butterfly per row
    if (lane == 0 || lane == 16) {               // C rows: low=r, high=r+8
      int base = ch * 16 + ((lane == 16) ? 8 : 0);
#pragma unroll
      for (int r = 0; r < 8; ++r) {
        int node = base + r;
        if (node < len) sc[node] = scc[r] + bb2;
      }
    }
  }
  __syncthreads();

  // ---- Phase 2: segment softmax (stable) -------------------------------
  float m1 = -3.402823466e38f;
  for (int i = tid; i < len; i += 256) m1 = fmaxf(m1, sc[i]);
  red[tid] = m1; __syncthreads();
  for (int s = 128; s > 0; s >>= 1) {
    if (tid < s) red[tid] = fmaxf(red[tid], red[tid + s]);
    __syncthreads();
  }
  float smax = red[0];
  __syncthreads();

  float s1 = 0.f;
  for (int i = tid; i < len; i += 256) {
    float e = __expf(sc[i] - smax);              // native v_exp_f32
    sc[i] = e;                                   // e_i kept in LDS
    s1 += e;
  }
  red[tid] = s1; __syncthreads();
  for (int s = 128; s > 0; s >>= 1) {
    if (tid < s) red[tid] += red[tid + s];
    __syncthreads();
  }
  float inv = (red[0] > 0.f) ? 1.f / red[0] : 0.f;
  __syncthreads();

  // ---- Phase 3: pooling (L2-hot re-read of the segment rows) -----------
  const int f = tid & 127;
  const int half = tid >> 7;
  float acc = 0.f, mx = -3.402823466e38f;
#pragma unroll 4
  for (int i = half; i < len; i += 2) {
    float hv = h[(size_t)(n0 + i) * HDIM + f];
    acc += sc[i] * hv;
    mx = fmaxf(mx, hv);
  }
  if (half == 0) { pa[f] = acc; pm[f] = mx; }
  __syncthreads();
  if (half == 1) {
    float fa = (pa[f] + acc) * inv;
    float fm = fmaxf(pm[f], mx);
    _Float16* gr = g16 + (size_t)b * 256;
    gr[f]       = (_Float16)fa;                  // attn-pool -> k = 0..127
    gr[128 + f] = (_Float16)fm;                  // max-pool  -> k = 128..255
  }
}

// ---------------------------------------------------------------------------
// Kernel 3: out = relu(g @ mW1 + mb1) @ mw2 + mb2.  One wave per 16-row tile.
// ---------------------------------------------------------------------------
__global__ __launch_bounds__(256, 4) void mlp_kernel(
    const _Float16* __restrict__ g16, const _Float16* __restrict__ mlpF,
    const float* __restrict__ mb1, const float* __restrict__ mw2,
    const float* __restrict__ mb2, float* __restrict__ out, int B) {
  const int tid = threadIdx.x;
  const int wave = tid >> 5, lane = tid & 31;
  const int lanelo = lane & 15;
  const int tile = blockIdx.x * 8 + wave;
  const int R = tile * 16;
  if (R >= B) return;                            // wave-uniform

  const int koff = (lane < 16) ? 0 : 8;
  const _Float16* rowp = g16 + (size_t)(R + lanelo) * 256;
  v16h A[8];
#pragma unroll
  for (int ks = 0; ks < 8; ++ks)
    A[ks] = load_frag(rowp + ks * 32 + koff);

  float part[8];
#pragma unroll
  for (int r = 0; r < 8; ++r) part[r] = 0.f;

#pragma unroll
  for (int nt = 0; nt < 8; ++nt) {
    float bias = mb1[nt * 16 + lanelo];
    v8f c;
#pragma unroll
    for (int r = 0; r < 8; ++r) c[r] = bias;
#pragma unroll
    for (int ks = 0; ks < 8; ++ks) {
      v16h Bf = *(const v16h*)(mlpF + ((size_t)((nt * 8 + ks) * 32 + lane)) * 16);
      c = __builtin_amdgcn_wmma_f32_16x16x32_f16(false, A[ks], false, Bf,
                                                 (short)0, c, false, false);
    }
    float w2c = mw2[nt * 16 + lanelo];
#pragma unroll
    for (int r = 0; r < 8; ++r)
      part[r] += fmaxf(c[r], 0.f) * w2c;         // fused ReLU + dot
  }
  float bb = mb2[0];
  float scc[8];
#pragma unroll
  for (int r = 0; r < 8; ++r)
    scc[r] = group_sum16(part[r]);
  if (lane == 0) {
#pragma unroll
    for (int r = 0; r < 8; ++r) out[R + r] = scc[r] + bb;
  }
  if (lane == 16) {
#pragma unroll
    for (int r = 0; r < 8; ++r) out[R + 8 + r] = scc[r] + bb;
  }
}

// ---------------------------------------------------------------------------
extern "C" void kernel_launch(void* const* d_in, const int* in_sizes, int n_in,
                              void* d_out, int out_size, void* d_ws, size_t ws_size,
                              hipStream_t stream) {
  (void)n_in; (void)ws_size;
  const float* h   = (const float*)d_in[0];
  const int*   bat = (const int*)d_in[1];
  const float* gw1 = (const float*)d_in[2];
  const float* gb1 = (const float*)d_in[3];
  const float* gw2 = (const float*)d_in[4];
  const float* gb2 = (const float*)d_in[5];
  const float* mw1 = (const float*)d_in[6];
  const float* mb1 = (const float*)d_in[7];
  const float* mw2 = (const float*)d_in[8];
  const float* mb2 = (const float*)d_in[9];
  float* out = (float*)d_out;

  const int N = in_sizes[1];        // node count
  const int B = out_size;           // NUM_OBS == 1 -> one output per segment

  // workspace layout: [gate frags 32KB][mlp frags 64KB][g16 (B*256 halves)]
  char* ws = (char*)d_ws;
  _Float16* gateF = (_Float16*)ws;
  _Float16* mlpF  = (_Float16*)(ws + 32768);
  _Float16* g16   = (_Float16*)(ws + 32768 + 65536);

  prep_frags<<<12, 256, 0, stream>>>(gw1, mw1, gateF, mlpF);
  seg_kernel<<<B, 256, 0, stream>>>(h, bat, gb1, gw2, gb2, gateF, g16, N);
  mlp_kernel<<<(B + 127) / 128, 256, 0, stream>>>(g16, mlpF, mb1, mw2, mb2, out, B);
}
